// EncoderDecoder_14851996909748
// MI455X (gfx1250) — matrix operand order repaired and checked by reading, not compiled, not run on hardware
//
#include <hip/hip_runtime.h>

// ---------- types ----------
typedef __bf16 bf16_t;
typedef __attribute__((ext_vector_type(16))) __bf16 v16bf;
typedef __attribute__((ext_vector_type(8)))  __bf16 v8bf;
typedef __attribute__((ext_vector_type(8)))  float  v8f;

#define HID   1024      // H
#define BATCH 512       // batch
#define G4H   4096      // 4*H
#define KT    (HID/32)  // 32 k-tiles of K=32 (bf16 WMMA depth)

// ============================================================================
// One-time: pack Whh [4H, H] fp32 row-major -> bf16 WMMA B-fragment layout.
// B (32x16 bf16, K x N): lanes 0-15 hold K=0..7,16..23 (slot = k%8 + 8*(k/16)), N = lane;
//                        lanes 16-31 hold K=8..15,24..31 (same slots), N = lane-16.
// Packed blocks: [n_tile][k_tile][lane][slot0..15], n_tile = j/16 over 4H.
// ============================================================================
__global__ void pack_whh_kernel(const float* __restrict__ W,
                                bf16_t* __restrict__ out) {
    int idx = blockIdx.x * blockDim.x + threadIdx.x;
    if (idx >= G4H * HID) return;
    int j = idx / HID;            // gate row (N dim over 4H)
    int k = idx % HID;            // K dim
    int nt = j >> 4, n = j & 15;
    int kt = k >> 5, kk = k & 31;
    int lane = n + ((kk & 8) ? 16 : 0);
    int slot = (kk & 7) + ((kk & 16) ? 8 : 0);
    out[(((size_t)nt * KT + kt) * 32 + lane) * 16 + slot] = (bf16_t)W[idx];
}

// ============================================================================
// One-time: zero h (bf16 row-major, buffer 0) and c.
// ============================================================================
__global__ void zero_state_kernel(bf16_t* __restrict__ hb0,
                                  float* __restrict__ c) {
    int i = blockIdx.x * blockDim.x + threadIdx.x;
    if (i < BATCH * HID) {
        hb0[i] = (bf16_t)0.0f;
        c[i]   = 0.0f;
    }
}

__device__ __forceinline__ float sigmoidf_(float x) {
    return 1.0f / (1.0f + __expf(-x));
}
__device__ __forceinline__ float tanhf_(float x) {
    float e = __expf(-2.0f * fabsf(x));      // in (0,1], no overflow
    float r = (1.0f - e) / (1.0f + e);
    return copysignf(r, x);
}

// ============================================================================
// Fully fused LSTM step. Each wave computes a 16(batch) x 16(k) tile of the
// state update: its 4 WMMA accumulators are the four gates i,f,g,o at the
// SAME (batch,k) tile (j = gate*H + k), so the whole pointwise update stays
// in registers. h is ping-ponged (read hbIn, write hbOut): no inter-block
// sync needed. A-fragments load straight from row-major bf16 h.
// k-loop: 2x-unrolled ping-pong double buffer -> loads for kt+2 issue before
// the WMMAs of kt+1, no register-rotation copies at the backedge.
// B-fragment addresses: one per-lane base + immediate offsets (gate stride
// 2 MB fits the signed 24-bit IOFFSET).
// 8 waves/WG (2 batch x 4 k) -> 32x64 tile; grid (16,16) = 256 WGs.
// ============================================================================
__global__ __launch_bounds__(256)
void lstm_step_fused_kernel(const bf16_t* __restrict__ hbIn,
                            bf16_t* __restrict__ hbOut,
                            const bf16_t* __restrict__ WP,   // packed 4H x H
                            const float*  __restrict__ bias, // [4H]
                            const float*  __restrict__ wih,  // [4H] (I==1)
                            const float*  __restrict__ xin,  // [BATCH]
                            float*        __restrict__ c,    // [BATCH,H]
                            float*        __restrict__ h32)  // [BATCH,H] or null
{
    const int lane  = threadIdx.x & 31;
    const int wave  = threadIdx.x >> 5;
    const int wm    = wave >> 2;          // 0..1 (batch)
    const int wn    = wave & 3;           // 0..3 (k)
    const int Mbase = blockIdx.x * 32 + wm * 16;
    const int Kbase = blockIdx.y * 64 + wn * 16;
    const int kc    = Kbase >> 4;         // k-tile index (16-wide)

    // Per-lane fragment base addresses (byte arithmetic, immediate offsets).
    // A row-major: lanes 0-15 row M=lane, lanes 16-31 row M=lane-16 (+8 elems);
    //   kt advance = 64 B; hi half at +32 B.
    // B packed: block stride per kt = 1024 B; gate stride = 64*KT*1024 = 2 MB.
    const char* aP = (const char*)hbIn +
        ((size_t)(Mbase + (lane & 15)) * HID + ((lane >= 16) ? 8 : 0)) * 2;
    const char* bP = (const char*)WP + (size_t)kc * (KT * 1024) + lane * 32;

    union AV { v16bf v; struct { v8bf lo, hi; } p; };

    v8f acc[4] = {v8f{}, v8f{}, v8f{}, v8f{}};

    AV a0, a1, b0[4], b1[4];

    auto loadA = [&](AV& A, int kt) {
        const char* p = aP + kt * 64;
        A.p.lo = *(const v8bf*)(p);
        A.p.hi = *(const v8bf*)(p + 32);
    };
    auto loadB = [&](AV* B, int kt) {
        const char* p = bP + kt * 1024;
        #pragma unroll
        for (int t = 0; t < 4; ++t) {
            B[t].p.lo = *(const v8bf*)(p + t * 2097152);
            B[t].p.hi = *(const v8bf*)(p + t * 2097152 + 16);
        }
    };
    auto mm4 = [&](AV& A, AV* B) {
        #pragma unroll
        for (int t = 0; t < 4; ++t)
            acc[t] = __builtin_amdgcn_wmma_f32_16x16x32_bf16(
                false, A.v, false, B[t].v, (short)0, acc[t], false, false);
    };

    // prologue: fill both buffers
    loadA(a0, 0); loadB(b0, 0);
    loadA(a1, 1); loadB(b1, 1);

    // steady state: consume one buffer, refill it two tiles ahead
    for (int kt = 0; kt < KT - 2; kt += 2) {
        mm4(a0, b0);
        loadA(a0, kt + 2); loadB(b0, kt + 2);
        mm4(a1, b1);
        loadA(a1, kt + 3); loadB(b1, kt + 3);
    }
    // epilogue pair (buffers already loaded)
    mm4(a0, b0);
    mm4(a1, b1);

    // ---- fused LSTM pointwise epilogue (all in registers) ----
    // C/D layout: VGPR r, lanes 0-15 -> (M=r, N=lane); lanes 16-31 -> (M=r+8).
    const int n    = lane & 15;
    const int k    = Kbase + n;
    const int xoff = Mbase + ((lane >= 16) ? 8 : 0);

    const float bi  = bias[k],           wi = wih[k];
    const float bf_ = bias[HID + k],     wf = wih[HID + k];
    const float bg  = bias[2 * HID + k], wg = wih[2 * HID + k];
    const float bo  = bias[3 * HID + k], wo = wih[3 * HID + k];

    #pragma unroll
    for (int r = 0; r < 8; ++r) {
        const int    bidx = xoff + r;
        const float  xv   = xin[bidx];
        const size_t off  = (size_t)bidx * HID + k;
        const float ig = sigmoidf_(acc[0][r] + bi  + xv * wi);
        const float fg = sigmoidf_(acc[1][r] + bf_ + xv * wf);
        const float gg = tanhf_   (acc[2][r] + bg  + xv * wg);
        const float og = sigmoidf_(acc[3][r] + bo  + xv * wo);
        const float cn = fg * c[off] + ig * gg;
        c[off] = cn;
        const float hn = og * tanhf_(cn);
        hbOut[off] = (bf16_t)hn;
        if (h32) h32[off] = hn;
    }
}

// ============================================================================
// Decoder linear: y[b] = dot(h32[b,:], linW) + linb; feeds back as next input.
// One wave per batch row; shuffle reduction. Grid 64 x 256 threads (8 waves).
// ============================================================================
__global__ void decoder_linear_kernel(const float* __restrict__ h32,
                                      const float* __restrict__ linW,
                                      const float* __restrict__ linb,
                                      float* __restrict__ yout,
                                      float* __restrict__ yin) {
    int wave = threadIdx.x >> 5;
    int lane = threadIdx.x & 31;
    int b = blockIdx.x * 8 + wave;
    const float* hr = h32 + (size_t)b * HID;
    float s = 0.0f;
    for (int k = lane; k < HID; k += 32) s += hr[k] * linW[k];
    #pragma unroll
    for (int off = 16; off; off >>= 1) s += __shfl_down(s, off, 32);
    if (lane == 0) {
        float y = s + linb[0];
        yout[b] = y;
        yin[b]  = y;
    }
}

// ============================================================================
// Host launcher
// ============================================================================
extern "C" void kernel_launch(void* const* d_in, const int* in_sizes, int n_in,
                              void* d_out, int out_size, void* d_ws, size_t ws_size,
                              hipStream_t stream) {
    const float* x       = (const float*)d_in[0];  // [512,512,1]
    const float* enc_Wih = (const float*)d_in[1];  // [4096]
    const float* enc_Whh = (const float*)d_in[2];  // [4096,1024]
    const float* enc_b   = (const float*)d_in[3];  // [4096]
    const float* dec_Wih = (const float*)d_in[4];
    const float* dec_Whh = (const float*)d_in[5];
    const float* dec_b   = (const float*)d_in[6];
    const float* lin_W   = (const float*)d_in[7];  // [1024]
    const float* lin_b   = (const float*)d_in[8];  // [1]
    float* out = (float*)d_out;                    // [64,512,1]

    const int SEQ_LEN = 512, TGT_LEN = 64;

    // Workspace layout (~22 MB, L2-resident)
    char* ws = (char*)d_ws;
    size_t off = 0;
    bf16_t* encWP = (bf16_t*)(ws + off); off += (size_t)G4H * HID * sizeof(bf16_t);   // 8 MB
    bf16_t* decWP = (bf16_t*)(ws + off); off += (size_t)G4H * HID * sizeof(bf16_t);   // 8 MB
    bf16_t* hb0   = (bf16_t*)(ws + off); off += (size_t)BATCH * HID * sizeof(bf16_t); // 1 MB
    bf16_t* hb1   = (bf16_t*)(ws + off); off += (size_t)BATCH * HID * sizeof(bf16_t); // 1 MB
    float*  c     = (float*) (ws + off); off += (size_t)BATCH * HID * sizeof(float);  // 2 MB
    float*  h32   = (float*) (ws + off); off += (size_t)BATCH * HID * sizeof(float);  // 2 MB
    float*  y_in  = (float*) (ws + off); off += (size_t)BATCH * sizeof(float);        // 2 KB

    // One-time weight packing + state zero (re-run every call: deterministic)
    {
        int nw = G4H * HID;
        int blocks = (nw + 255) / 256;
        pack_whh_kernel<<<blocks, 256, 0, stream>>>(enc_Whh, encWP);
        pack_whh_kernel<<<blocks, 256, 0, stream>>>(dec_Whh, decWP);
        int m = BATCH * HID;
        zero_state_kernel<<<(m + 255) / 256, 256, 0, stream>>>(hb0, c);
    }

    bf16_t* hb[2] = {hb0, hb1};
    dim3 grid(BATCH / 32, HID / 64);   // 16 x 16 = 256 WGs

    // Encoder: 512 dependent fused steps
    for (int t = 0; t < SEQ_LEN; ++t) {
        lstm_step_fused_kernel<<<grid, 256, 0, stream>>>(
            hb[t & 1], hb[(t + 1) & 1], encWP, enc_b, enc_Wih,
            x + (size_t)t * BATCH, c, nullptr);
    }

    // Decoder: 64 dependent fused steps + feedback linear
    for (int t = 0; t < TGT_LEN; ++t) {
        const int s = SEQ_LEN + t;
        const float* xinp = (t == 0) ? (x + (size_t)(SEQ_LEN - 1) * BATCH) : y_in;
        lstm_step_fused_kernel<<<grid, 256, 0, stream>>>(
            hb[s & 1], hb[(s + 1) & 1], decWP, dec_b, dec_Wih, xinp, c, h32);
        decoder_linear_kernel<<<BATCH / 8, 256, 0, stream>>>(
            h32, lin_W, lin_b, out + (size_t)t * BATCH, y_in);
    }
}
